// FTTransformerPNAParallelLayer_27857157882091
// MI455X (gfx1250) — compile-verified, matrix operand order checked by reading, not run on hardware
//
#include <hip/hip_runtime.h>

// ---------------------------------------------------------------------------
// MI455X (gfx1250) implementation: f16 WMMA GEMMs (f32 accumulate), fused
// edge-gather A-operand, fused PNA aggregation epilogue (atomics),
// double-buffered LDS-staged weight tiles (one barrier per K-step),
// 8 v_wmma_f32_16x16x32_f16 per staged tile per wave.
// ---------------------------------------------------------------------------

typedef __attribute__((ext_vector_type(16))) _Float16 v16h;
typedef __attribute__((ext_vector_type(8)))  _Float16 v8h;
typedef __attribute__((ext_vector_type(8)))  float    v8f;

constexpr int  NN  = 20000;
constexpr int  NE  = 320000;
constexpr int  BR  = 4096, NC = 32;
constexpr long TOK = (long)BR * NC;   // 131072 tokens

// ---- order-preserving float<->uint key for atomic max/min -----------------
__device__ __forceinline__ unsigned fkey(float f) {
  unsigned b = __float_as_uint(f);
  return (b & 0x80000000u) ? ~b : (b | 0x80000000u);
}
__device__ __forceinline__ float fkeyinv(unsigned k) {
  unsigned b = (k & 0x80000000u) ? (k & 0x7fffffffu) : ~k;
  return __uint_as_float(b);
}

// ---------------------------------------------------------------------------
// Generic WMMA GEMM:  Out[M,N] = epi( A[M,K] * W[N,K]^T + bias )
//   concat mode: A row e = [ N0[I0[e]] (128) | N1[I1[e]] (128) | E2[e] (128) ]
//   epilogues: relu, f16 store, f32 store (+resid*scale), PNA agg atomics
// Block: 128 threads = 4 waves; tile 128(M) x 64(N); wave = 32 rows.
// ---------------------------------------------------------------------------
struct GemmCfg {
  const _Float16* A;
  const _Float16* N0; const int* I0;
  const _Float16* N1; const int* I1;
  const _Float16* E2;
  const _Float16* W;
  const float* bias;
  const float* resid;
  float resid_scale;
  int relu;
  _Float16* outH;
  float*    outF;
  float* aggSum; float* aggSsq;
  unsigned* aggMaxK; unsigned* aggMinK;
  const int* aggIdx;
  int M, N, K;
  int concat;
};

__launch_bounds__(128)
__global__ void k_gemm(GemmCfg c) {
  // double-buffered 64x32 weight tiles (2 x 4 KB)
  __shared__ __align__(16) _Float16 ldsW[2][64 * 32];

  const int lane  = threadIdx.x & 31;
  const int wave  = threadIdx.x >> 5;
  const int nBase = blockIdx.x * 64;
  const int mBase = blockIdx.y * 128 + wave * 32;   // 32 rows per wave
  const int ml = lane & 15;          // row/col within 16
  const int hs = lane >> 4;          // K-half select (ISA 16-bit A layout)

  // Two 16-row A fragments per wave.
  int rowc[2];
  const _Float16* aRow[2];
  const _Float16* p0[2]; const _Float16* p1[2]; const _Float16* p2[2];
#pragma unroll
  for (int t = 0; t < 2; ++t) {
    const int row = mBase + t * 16 + ml;
    rowc[t] = (row < c.M) ? row : (c.M - 1);
    if (c.concat) {
      p0[t] = c.N0 + (long)c.I0[rowc[t]] * 128;
      p1[t] = c.N1 + (long)c.I1[rowc[t]] * 128;
      p2[t] = c.E2 + (long)rowc[t] * 128;
      aRow[t] = nullptr;
    } else {
      aRow[t] = c.A + (long)rowc[t] * c.K;
      p0[t] = p1[t] = p2[t] = nullptr;
    }
  }

  auto stage = [&](int buf, int k) {
    const int nl   = threadIdx.x >> 1;
    const int half = (threadIdx.x & 1) * 16;
    const _Float16* wsrc = c.W + (long)(nBase + nl) * c.K + k + half;
    *(v8h*)&ldsW[buf][nl * 32 + half]     = *(const v8h*)(wsrc);
    *(v8h*)&ldsW[buf][nl * 32 + half + 8] = *(const v8h*)(wsrc + 8);
  };

  v8f acc[2][4];
#pragma unroll
  for (int t = 0; t < 2; ++t)
#pragma unroll
    for (int j = 0; j < 4; ++j) acc[t][j] = v8f{};

  stage(0, 0);
  int p = 0;
  for (int k = 0; k < c.K; k += 32, p ^= 1) {
    // Single barrier per step: publishes stage of ldsW[p] (written last
    // iteration) and retires last iteration's reads of ldsW[p^1], which we
    // are about to overwrite with the next tile.
    __syncthreads();
    if (k + 32 < c.K) stage(p ^ 1, k + 32);

    // A fragments (16x32 each): lane hs picks K 0-7/16-23 vs 8-15/24-31
    union { v16h v; v8h h[2]; } af[2];
#pragma unroll
    for (int t = 0; t < 2; ++t) {
      const _Float16* ap; int koff;
      if (c.concat) {
        if      (k < 128) { ap = p0[t]; koff = k;       }
        else if (k < 256) { ap = p1[t]; koff = k - 128; }
        else              { ap = p2[t]; koff = k - 256; }
      } else { ap = aRow[t]; koff = k; }
      const int kb = koff + hs * 8;
      af[t].h[0] = *(const v8h*)(ap + kb);
      af[t].h[1] = *(const v8h*)(ap + kb + 16);
    }

#pragma unroll
    for (int j = 0; j < 4; ++j) {
      union { v16h v; v8h h[2]; } bf;
      const int nl = j * 16 + ml;
      bf.h[0] = *(const v8h*)&ldsW[p][nl * 32 + hs * 8];
      bf.h[1] = *(const v8h*)&ldsW[p][nl * 32 + hs * 8 + 16];
      acc[0][j] = __builtin_amdgcn_wmma_f32_16x16x32_f16(
          false, af[0].v, false, bf.v, (short)0, acc[0][j], false, false);
      acc[1][j] = __builtin_amdgcn_wmma_f32_16x16x32_f16(
          false, af[1].v, false, bf.v, (short)0, acc[1][j], false, false);
    }
  }

  // Epilogue. C layout: VGPR r, lanes 0-15 -> M=r, lanes 16-31 -> M=r+8.
#pragma unroll
  for (int t = 0; t < 2; ++t) {
#pragma unroll
    for (int j = 0; j < 4; ++j) {
      const int n  = nBase + j * 16 + ml;
      const float bv = c.bias[n];
#pragma unroll
      for (int r = 0; r < 8; ++r) {
        const int m = mBase + t * 16 + hs * 8 + r;
        if (m >= c.M) continue;
        float v = acc[t][j][r] + bv;
        if (c.relu) v = v > 0.f ? v : 0.f;
        if (c.aggSum) {                       // PNA: mean/max/min/std feed
          const int dnode = c.aggIdx[m];
          const long o = (long)dnode * c.N + n;
          atomicAdd(&c.aggSum[o], v);
          atomicAdd(&c.aggSsq[o], v * v);
          atomicMax(&c.aggMaxK[o], fkey(v));
          atomicMin(&c.aggMinK[o], fkey(v));
        } else {
          const long o = (long)m * c.N + n;
          if (c.outH) c.outH[o] = (_Float16)v;
          if (c.outF) {
            float ov = v * c.resid_scale;
            if (c.resid) ov += c.resid[o];
            c.outF[o] = ov;
          }
        }
      }
    }
  }
}

// ---------------------------------------------------------------------------
// Per-batch MHA (S=32, 8 heads, hd=16) — tiny FLOPs, LDS + VALU.
// qkv rows: [q(128)|k(128)|v(128)] per token. One block per batch element.
// ---------------------------------------------------------------------------
__launch_bounds__(256)
__global__ void k_attention(const _Float16* __restrict__ qkv, _Float16* __restrict__ out) {
  __shared__ __align__(16) _Float16 sm[32 * 384];   // 24 KB
  const long b = blockIdx.x;
  const unsigned* s32 = (const unsigned*)(qkv + b * 32 * 384);
  unsigned* l32 = (unsigned*)sm;
  for (int i = threadIdx.x; i < 32 * 384 / 2; i += 256) l32[i] = s32[i];
  __syncthreads();

  const int h = threadIdx.x >> 5;   // head
  const int q = threadIdx.x & 31;   // query token
  float qv[16];
#pragma unroll
  for (int i = 0; i < 16; ++i) qv[i] = (float)sm[q * 384 + h * 16 + i];

  float sc[32]; float mx = -1e30f;
  for (int j = 0; j < 32; ++j) {
    float d = 0.f;
#pragma unroll
    for (int i = 0; i < 16; ++i) d += qv[i] * (float)sm[j * 384 + 128 + h * 16 + i];
    d *= 0.25f;                      // 1/sqrt(16)
    sc[j] = d; mx = fmaxf(mx, d);
  }
  float den = 0.f;
  for (int j = 0; j < 32; ++j) { sc[j] = expf(sc[j] - mx); den += sc[j]; }
  const float inv = 1.f / den;
  float o[16];
#pragma unroll
  for (int i = 0; i < 16; ++i) o[i] = 0.f;
  for (int j = 0; j < 32; ++j) {
    const float p = sc[j] * inv;
#pragma unroll
    for (int i = 0; i < 16; ++i) o[i] += p * (float)sm[j * 384 + 256 + h * 16 + i];
  }
  _Float16* dst = out + b * 32 * 128 + q * 128 + h * 16;
#pragma unroll
  for (int i = 0; i < 16; ++i) dst[i] = (_Float16)o[i];
}

// ---------------------------------------------------------------------------
// Row LayerNorm over D=128 (one block per row)
// ---------------------------------------------------------------------------
__launch_bounds__(128)
__global__ void k_layernorm(const float* __restrict__ x, float* outF, _Float16* outH,
                            const float* g, const float* b) {
  __shared__ float red[128];
  const long row = blockIdx.x; const int t = threadIdx.x;
  const float v = x[row * 128 + t];
  red[t] = v; __syncthreads();
  for (int s = 64; s > 0; s >>= 1) { if (t < s) red[t] += red[t + s]; __syncthreads(); }
  const float mean = red[0] * (1.f / 128.f); __syncthreads();
  const float d = v - mean;
  red[t] = d * d; __syncthreads();
  for (int s = 64; s > 0; s >>= 1) { if (t < s) red[t] += red[t + s]; __syncthreads(); }
  const float o = d * rsqrtf(red[0] * (1.f / 128.f) + 1e-5f) * g[t] + b[t];
  if (outF) outF[row * 128 + t] = o;
  if (outH) outH[row * 128 + t] = (_Float16)o;
}

// Two stacked LayerNorms in-place (post-norm ln2 then outer tabn)
__launch_bounds__(128)
__global__ void k_double_ln(float* y, const float* g1, const float* b1,
                            const float* g2, const float* b2) {
  __shared__ float red[128];
  const long row = blockIdx.x; const int t = threadIdx.x;
  float o = y[row * 128 + t];
#pragma unroll
  for (int pass = 0; pass < 2; ++pass) {
    const float* g = pass ? g2 : g1;
    const float* b = pass ? b2 : b1;
    red[t] = o; __syncthreads();
    for (int s = 64; s > 0; s >>= 1) { if (t < s) red[t] += red[t + s]; __syncthreads(); }
    const float mean = red[0] * (1.f / 128.f); __syncthreads();
    const float d = o - mean;
    red[t] = d * d; __syncthreads();
    for (int s = 64; s > 0; s >>= 1) { if (t < s) red[t] += red[t + s]; __syncthreads(); }
    o = d * rsqrtf(red[0] * (1.f / 128.f) + 1e-5f) * g[t] + b[t];
    __syncthreads();
  }
  y[row * 128 + t] = o;
}

// ---------------------------------------------------------------------------
// Small helpers
// ---------------------------------------------------------------------------
__global__ void k_cvt(const float* __restrict__ s, _Float16* __restrict__ d, long n) {
  const long i = (long)blockIdx.x * blockDim.x + threadIdx.x;
  if (i < n) d[i] = (_Float16)s[i];
}

__global__ void k_init_agg(float* sum, float* ssq, unsigned* mxk, unsigned* mnk,
                           float* cnt, float* avgl) {
  const long i = (long)blockIdx.x * blockDim.x + threadIdx.x;
  if (i < (long)NN * 128) {
    sum[i] = 0.f; ssq[i] = 0.f;
    mxk[i] = 0x007fffffu;           // fkey(-inf)
    mnk[i] = 0xff800000u;           // fkey(+inf)
  }
  if (i < NN) cnt[i] = 0.f;
  if (i == 0) avgl[0] = 0.f;
}

__global__ void k_count(const int* __restrict__ ei, float* cnt) {
  const int e = blockIdx.x * blockDim.x + threadIdx.x;
  if (e < NE) atomicAdd(&cnt[ei[NE + e]], 1.0f);    // dst = edge_index[1]
}

__launch_bounds__(256)
__global__ void k_avglog(const float* __restrict__ cnt, float* avgl) {
  __shared__ float s[256];
  float a = 0.f;
  for (int i = blockIdx.x * 256 + threadIdx.x; i < NN; i += gridDim.x * 256)
    a += log1pf(cnt[i]);
  s[threadIdx.x] = a; __syncthreads();
  for (int st = 128; st > 0; st >>= 1) {
    if (threadIdx.x < st) s[threadIdx.x] += s[threadIdx.x + st];
    __syncthreads();
  }
  if (threadIdx.x == 0) atomicAdd(avgl, s[0]);
}

// Build out13 = [x | mean,mx,mn,std | agg*logd/al | agg*al/logd]  (f16, 1664 wide)
__global__ void k_assemble(const float* __restrict__ xg, const float* __restrict__ sum,
                           const float* __restrict__ ssq, const unsigned* __restrict__ mxk,
                           const unsigned* __restrict__ mnk, const float* __restrict__ cnt,
                           const float* __restrict__ avgl, _Float16* __restrict__ out13) {
  const long i = (long)blockIdx.x * blockDim.x + threadIdx.x;
  if (i >= (long)NN * 128) return;
  const long d = i >> 7; const int f = (int)(i & 127);
  const float c    = cnt[d];
  const float safe = fmaxf(c, 1.f);
  const float mean = sum[i] / safe;
  float var = ssq[i] / safe - mean * mean;
  var = var > 0.f ? var : 0.f;
  const float sd = sqrtf(var + 1e-5f);
  const bool  has = c > 0.f;
  const float mx = has ? fkeyinv(mxk[i]) : 0.f;
  const float mn = has ? fkeyinv(mnk[i]) : 0.f;
  const float al   = avgl[0] * (1.f / (float)NN);
  const float logd = log1pf(fmaxf(c, 1.f));
  const float samp = logd / al, satt = al / logd;
  _Float16* o = out13 + d * 1664;
  o[f]        = (_Float16)xg[i];
  o[128 + f]  = (_Float16)mean;        o[256 + f]  = (_Float16)mx;
  o[384 + f]  = (_Float16)mn;          o[512 + f]  = (_Float16)sd;
  o[640 + f]  = (_Float16)(mean*samp); o[768 + f]  = (_Float16)(mx*samp);
  o[896 + f]  = (_Float16)(mn*samp);   o[1024 + f] = (_Float16)(sd*samp);
  o[1152 + f] = (_Float16)(mean*satt); o[1280 + f] = (_Float16)(mx*satt);
  o[1408 + f] = (_Float16)(mn*satt);   o[1536 + f] = (_Float16)(sd*satt);
}

// BatchNorm batch statistics per feature (128 blocks, one per feature)
__launch_bounds__(256)
__global__ void k_bnstats(const float* __restrict__ conv, float* bnst) {
  __shared__ float s1[256], s2[256];
  const int f = blockIdx.x;
  float a = 0.f, b = 0.f;
  for (int i = threadIdx.x; i < NN; i += 256) {
    const float v = conv[(long)i * 128 + f];
    a += v; b += v * v;
  }
  s1[threadIdx.x] = a; s2[threadIdx.x] = b; __syncthreads();
  for (int s = 128; s > 0; s >>= 1) {
    if (threadIdx.x < s) { s1[threadIdx.x] += s1[threadIdx.x + s]; s2[threadIdx.x] += s2[threadIdx.x + s]; }
    __syncthreads();
  }
  if (threadIdx.x == 0) { bnst[f] = s1[0]; bnst[128 + f] = s2[0]; }
}

// x_new = (x_gnn + relu(BN(conv)))/2 ; conv (=d_out region1) updated in place
__global__ void k_combine(const float* __restrict__ xg, float* conv,
                          const float* __restrict__ bnst,
                          const float* __restrict__ g, const float* __restrict__ b,
                          _Float16* __restrict__ xnh) {
  const long i = (long)blockIdx.x * blockDim.x + threadIdx.x;
  if (i >= (long)NN * 128) return;
  const int f = (int)(i & 127);
  const float m   = bnst[f] * (1.f / (float)NN);
  const float var = bnst[128 + f] * (1.f / (float)NN) - m * m;
  float bn = (conv[i] - m) * rsqrtf(var + 1e-5f) * g[f] + b[f];
  bn = bn > 0.f ? bn : 0.f;
  const float xn = (xg[i] + bn) * 0.5f;
  conv[i] = xn;
  xnh[i]  = (_Float16)xn;
}

// ---------------------------------------------------------------------------
// Workspace layout (bytes). Persistent f16 weights, then a phase-overlaid
// arena (transformer then PNA). Peak requirement ~236 MB.
// ---------------------------------------------------------------------------
constexpr long OFF_WQKVH = 0;
constexpr long OFF_WOH   = 98304;
constexpr long OFF_WF1H  = 131072;
constexpr long OFF_WF2H  = 262144;
constexpr long OFF_WEH   = 393216;
constexpr long OFF_WPREH = 425984;
constexpr long OFF_WPOSTH= 524288;
constexpr long OFF_WLINH = 950272;
constexpr long OFF_WEU1H = 983040;
constexpr long OFF_WEU2H = 1081344;
constexpr long ARENA     = 1114112;
// transformer phase
constexpr long R_XH   = ARENA + 0;            // 33,554,432  f16 x_tab
constexpr long R_QKVH = ARENA + 33554432;     // 100,663,296 f16 qkv
constexpr long R_AOH  = ARENA + 0;            // f16 attn out (reuse XH)
constexpr long R_Y1N  = ARENA + 33554432;     // 67,108,864  f32 ln1 out (reuse QKVH)
constexpr long R_Y1NH = ARENA + 0;            // f16 ln1 out (reuse AOH)
constexpr long R_TH   = ARENA + 100663296;    // 134,217,728 f16 ffn hidden
// PNA phase (transformer arena dead)
constexpr long R_EAH   = ARENA + 0;           // 81,920,000  f16 edge_attr
constexpr long R_XGH   = ARENA + 81920000;    // 5,120,000   f16 x_gnn
constexpr long R_EH    = ARENA + 87040000;    // 81,920,000  f16 encoded edges
constexpr long R_SUM   = ARENA + 168960000;   // 10,240,000  f32
constexpr long R_SSQ   = ARENA + 179200000;   // 10,240,000  f32
constexpr long R_MXK   = ARENA + 189440000;   // 10,240,000  u32 keys
constexpr long R_MNK   = ARENA + 199680000;   // 10,240,000  u32 keys
constexpr long R_CNT   = ARENA + 209920000;   // 80,000      f32
constexpr long R_AVGL  = ARENA + 210000000;   // 64          f32
constexpr long R_BNST  = ARENA + 210000064;   // 1,024       f32
constexpr long R_OUT13H= ARENA + 87040000;    // 66,560,000  f16 (reuse EH)
constexpr long R_CONV1H= ARENA + 153600000;   // 5,120,000   f16
constexpr long R_XNH   = ARENA + 81920000;    // 5,120,000   f16 (reuse XGH)
constexpr long R_T1H   = ARENA + 87040000;    // 81,920,000  f16 (reuse EH/OUT13)

extern "C" void kernel_launch(void* const* d_in, const int* in_sizes, int n_in,
                              void* d_out, int out_size, void* d_ws, size_t ws_size,
                              hipStream_t stream) {
  (void)in_sizes; (void)n_in; (void)out_size; (void)ws_size;

  const float* x_tab = (const float*)d_in[0];
  const float* x_gnn = (const float*)d_in[1];
  const int*   ei    = (const int*)d_in[2];     // [2, NE]: src row 0, dst row 1
  const float* eattr = (const float*)d_in[3];
  const float* Wqkv = (const float*)d_in[4];  const float* bqkv = (const float*)d_in[5];
  const float* Wo   = (const float*)d_in[6];  const float* bo   = (const float*)d_in[7];
  const float* ln1g = (const float*)d_in[8];  const float* ln1b = (const float*)d_in[9];
  const float* Wf1  = (const float*)d_in[10]; const float* bf1  = (const float*)d_in[11];
  const float* Wf2  = (const float*)d_in[12]; const float* bf2  = (const float*)d_in[13];
  const float* ln2g = (const float*)d_in[14]; const float* ln2b = (const float*)d_in[15];
  const float* tabg = (const float*)d_in[16]; const float* tabb = (const float*)d_in[17];
  const float* We   = (const float*)d_in[18]; const float* be   = (const float*)d_in[19];
  const float* Wpre = (const float*)d_in[20]; const float* bpre = (const float*)d_in[21];
  const float* Wpost= (const float*)d_in[22]; const float* bpost= (const float*)d_in[23];
  const float* Wlin = (const float*)d_in[24]; const float* blin = (const float*)d_in[25];
  const float* bng  = (const float*)d_in[26]; const float* bnb  = (const float*)d_in[27];
  const float* Weu1 = (const float*)d_in[28]; const float* beu1 = (const float*)d_in[29];
  const float* Weu2 = (const float*)d_in[30]; const float* beu2 = (const float*)d_in[31];

  char* ws = (char*)d_ws;
  auto Hp = [&](long o) { return (_Float16*)(ws + o); };
  auto Fp = [&](long o) { return (float*)(ws + o); };
  auto Up = [&](long o) { return (unsigned*)(ws + o); };

  float* out0 = (float*)d_out;                 // x_tab result [131072,128]
  float* out1 = out0 + TOK * 128;              // x_gnn result [20000,128]
  float* out2 = out1 + (long)NN * 128;         // edge result  [320000,128]

  auto cvt = [&](const float* s, _Float16* d, long n) {
    k_cvt<<<dim3((unsigned)((n + 255) / 256)), dim3(256), 0, stream>>>(s, d, n);
  };
  auto gemm = [&](const GemmCfg& c) {
    dim3 g((unsigned)(c.N / 64), (unsigned)((c.M + 127) / 128));
    k_gemm<<<g, dim3(128), 0, stream>>>(c);
  };

  // ---- weights -> f16 (persistent) ----
  cvt(Wqkv, Hp(OFF_WQKVH), 384 * 128);
  cvt(Wo,   Hp(OFF_WOH),   128 * 128);
  cvt(Wf1,  Hp(OFF_WF1H),  512 * 128);
  cvt(Wf2,  Hp(OFF_WF2H),  128 * 512);
  cvt(We,   Hp(OFF_WEH),   128 * 128);
  cvt(Wpre, Hp(OFF_WPREH), 128 * 384);
  cvt(Wpost,Hp(OFF_WPOSTH),128 * 1664);
  cvt(Wlin, Hp(OFF_WLINH), 128 * 128);
  cvt(Weu1, Hp(OFF_WEU1H), 128 * 384);
  cvt(Weu2, Hp(OFF_WEU2H), 128 * 128);

  // ================= Transformer branch =================
  cvt(x_tab, Hp(R_XH), TOK * 128);
  { GemmCfg c{}; c.A = Hp(R_XH); c.W = Hp(OFF_WQKVH); c.bias = bqkv;
    c.M = (int)TOK; c.N = 384; c.K = 128; c.outH = Hp(R_QKVH); gemm(c); }

  k_attention<<<dim3(BR), dim3(256), 0, stream>>>(Hp(R_QKVH), Hp(R_AOH));

  { GemmCfg c{}; c.A = Hp(R_AOH); c.W = Hp(OFF_WOH); c.bias = bo;
    c.M = (int)TOK; c.N = 128; c.K = 128;
    c.outF = out0; c.resid = x_tab; c.resid_scale = 1.0f; gemm(c); }  // y1 = x + o@Wo+bo

  k_layernorm<<<dim3((unsigned)TOK), dim3(128), 0, stream>>>(out0, Fp(R_Y1N), Hp(R_Y1NH), ln1g, ln1b);

  { GemmCfg c{}; c.A = Hp(R_Y1NH); c.W = Hp(OFF_WF1H); c.bias = bf1;
    c.M = (int)TOK; c.N = 512; c.K = 128; c.relu = 1; c.outH = Hp(R_TH); gemm(c); }

  { GemmCfg c{}; c.A = Hp(R_TH); c.W = Hp(OFF_WF2H); c.bias = bf2;
    c.M = (int)TOK; c.N = 128; c.K = 512;
    c.outF = out0; c.resid = Fp(R_Y1N); c.resid_scale = 1.0f; gemm(c); } // y2 = y1n + ff

  k_double_ln<<<dim3((unsigned)TOK), dim3(128), 0, stream>>>(out0, ln2g, ln2b, tabg, tabb);

  // ================= PNA branch =================
  cvt(eattr, Hp(R_EAH), (long)NE * 128);
  cvt(x_gnn, Hp(R_XGH), (long)NN * 128);

  { GemmCfg c{}; c.A = Hp(R_EAH); c.W = Hp(OFF_WEH); c.bias = be;
    c.M = NE; c.N = 128; c.K = 128; c.outH = Hp(R_EH); gemm(c); }       // edge encoder

  k_init_agg<<<dim3((NN * 128 + 255) / 256), dim3(256), 0, stream>>>(
      Fp(R_SUM), Fp(R_SSQ), Up(R_MXK), Up(R_MNK), Fp(R_CNT), Fp(R_AVGL));
  k_count<<<dim3((NE + 255) / 256), dim3(256), 0, stream>>>(ei, Fp(R_CNT));

  { GemmCfg c{}; c.concat = 1;                                           // m = h@Wpre+bpre
    c.N0 = Hp(R_XGH); c.I0 = ei + NE;   // x[dst]
    c.N1 = Hp(R_XGH); c.I1 = ei;        // x[src]
    c.E2 = Hp(R_EH);
    c.W = Hp(OFF_WPREH); c.bias = bpre; c.M = NE; c.N = 128; c.K = 384;
    c.aggSum = Fp(R_SUM); c.aggSsq = Fp(R_SSQ);
    c.aggMaxK = Up(R_MXK); c.aggMinK = Up(R_MNK); c.aggIdx = ei + NE; gemm(c); }

  k_avglog<<<dim3(64), dim3(256), 0, stream>>>(Fp(R_CNT), Fp(R_AVGL));
  k_assemble<<<dim3((NN * 128 + 255) / 256), dim3(256), 0, stream>>>(
      x_gnn, Fp(R_SUM), Fp(R_SSQ), Up(R_MXK), Up(R_MNK), Fp(R_CNT), Fp(R_AVGL), Hp(R_OUT13H));

  { GemmCfg c{}; c.A = Hp(R_OUT13H); c.W = Hp(OFF_WPOSTH); c.bias = bpost;
    c.M = NN; c.N = 128; c.K = 1664; c.outH = Hp(R_CONV1H); gemm(c); }

  { GemmCfg c{}; c.A = Hp(R_CONV1H); c.W = Hp(OFF_WLINH); c.bias = blin;
    c.M = NN; c.N = 128; c.K = 128; c.outF = out1; c.resid_scale = 1.0f; gemm(c); } // conv

  k_bnstats<<<dim3(128), dim3(256), 0, stream>>>(out1, Fp(R_BNST));
  k_combine<<<dim3((NN * 128 + 255) / 256), dim3(256), 0, stream>>>(
      x_gnn, out1, Fp(R_BNST), bng, bnb, Hp(R_XNH));

  { GemmCfg c{}; c.concat = 1;                                           // edge update MLP (1)
    c.N0 = Hp(R_XNH); c.I0 = ei;        // x_new[src]
    c.N1 = Hp(R_XNH); c.I1 = ei + NE;   // x_new[dst]
    c.E2 = Hp(R_EAH);
    c.W = Hp(OFF_WEU1H); c.bias = beu1; c.M = NE; c.N = 128; c.K = 384;
    c.relu = 1; c.outH = Hp(R_T1H); gemm(c); }

  { GemmCfg c{}; c.A = Hp(R_T1H); c.W = Hp(OFF_WEU2H); c.bias = beu2;    // edge update MLP (2)
    c.M = NE; c.N = 128; c.K = 128;
    c.outF = out2; c.resid = eattr; c.resid_scale = 0.5f; gemm(c); }     // eattr + upd/2
}
